// MPNN_63436666962551
// MI455X (gfx1250) — matrix operand order repaired and checked by reading, not compiled, not run on hardware
//
#include <hip/hip_runtime.h>

// GCN 3-layer MLP, N=100000 nodes, D=512, E=3.2M edges.
// GEMMs run on v_wmma_f32_16x16x32_bf16 (f32 accum), software-pipelined,
// with the activation tile staged in LDS (async global->LDS when available).

#define DD 512
#define BN_EPS 1e-5f

// Async global->LDS copy path (gfx1250) if the toolchain exposes the builtin.
#if defined(__has_builtin)
#if __has_builtin(__builtin_amdgcn_global_load_async_to_lds_b128) && \
    __has_builtin(__builtin_amdgcn_s_wait_asynccnt)
#define ASYNC_LDS 1
#endif
#endif
#ifndef ASYNC_LDS
#define ASYNC_LDS 0
#endif

typedef __attribute__((ext_vector_type(16))) __bf16 v16bf;
typedef __attribute__((ext_vector_type(8)))  __bf16 v8bf;
typedef __attribute__((ext_vector_type(8)))  float  v8f;
typedef int v4i_ __attribute__((vector_size(16)));   // matches builtin param type

static __device__ __forceinline__ unsigned short f2bf(float f) {
    // round-to-nearest-even f32 -> bf16
    unsigned int u = __float_as_uint(f);
    unsigned int r = (u + 0x7fffu + ((u >> 16) & 1u)) >> 16;
    return (unsigned short)r;
}

// ---------------- graph scale kernels ----------------

__global__ void init_deg_s(float* __restrict__ deg, float* __restrict__ s, int n) {
    int i = blockIdx.x * blockDim.x + threadIdx.x;
    if (i < n) { deg[i] = 1.0f; s[i] = 0.0f; }
}

__global__ void deg_kernel(const int* __restrict__ ei, int e, float* __restrict__ deg) {
    int i = blockIdx.x * blockDim.x + threadIdx.x;
    if (i >= e) return;
    int src = ei[i], dst = ei[e + i];
    if (src != dst) atomicAdd(&deg[dst], 1.0f);
}

__global__ void wsum_kernel(const int* __restrict__ ei, int e,
                            const float* __restrict__ deg, float* __restrict__ s) {
    int i = blockIdx.x * blockDim.x + threadIdx.x;
    if (i >= e) return;
    int src = ei[i], dst = ei[e + i];
    if (src != dst) atomicAdd(&s[src], rsqrtf(deg[src] * deg[dst]));
}

__global__ void s_final(float* __restrict__ s, const float* __restrict__ deg, int n) {
    int i = blockIdx.x * blockDim.x + threadIdx.x;
    if (i < n) s[i] += 1.0f / deg[i];
}

// ---------------- dtype conversion ----------------

__global__ void cvt_bf16(const float* __restrict__ in, unsigned short* __restrict__ out,
                         size_t count) {
    size_t i = (size_t)blockIdx.x * blockDim.x + threadIdx.x;
    if (i < count) out[i] = f2bf(in[i]);
}

// ---------------- WMMA GEMM + epilogue ----------------
// Y[m, o] = (sum_k Xb[m,k] * Wb[o,k]) * s[m] + bias[o]
// grid = (n/32, 2), block = 256 threads = 8 waves.
// Block tile: 32 rows x 256 cols. Waves: 2 row-tiles x 4 col-groups (64 cols).
// X tile (32x512 bf16, 32 KB) staged in LDS; W fragments streamed from L2.
// Register double-buffered K pipeline (16 steps of 16x16x32 WMMA).
__global__ void __launch_bounds__(256)
gcn_gemm_wmma(const unsigned short* __restrict__ Xb,   // [n, 512] bf16
              const unsigned short* __restrict__ Wb,   // [512, 512] bf16 (out, in)
              const float* __restrict__ bias,          // [512]
              const float* __restrict__ s,             // [n]
              float* __restrict__ Y,                   // [n, 512]
              float* __restrict__ colsum,
              float* __restrict__ colsq,
              int do_stats)
{
    __shared__ __align__(16) unsigned short atile[32][DD + 8];  // padded rows
    __shared__ float ls[256];
    __shared__ float lq[256];

    const int tid  = threadIdx.x;
    const long m0  = (long)blockIdx.x * 32;
    const int coff = blockIdx.y * 256;

    // ---- stage 32x512 bf16 X tile into LDS (2048 x 16B chunks, 8/thread) ----
#pragma unroll
    for (int c = 0; c < 8; ++c) {
        const int chunk = tid + c * 256;      // 0..2047
        const int row   = chunk >> 6;         // 64 x 16B chunks per row
        const int col8  = chunk & 63;
        const unsigned short* g = Xb + (m0 + row) * (size_t)DD + col8 * 8;
        unsigned short*       l = &atile[row][col8 * 8];
#if ASYNC_LDS
        __builtin_amdgcn_global_load_async_to_lds_b128(
            (__attribute__((address_space(1))) v4i_*)g,
            (__attribute__((address_space(3))) v4i_*)l, 0, 0);
#else
        *(uint4*)l = *(const uint4*)g;
#endif
    }
    if (do_stats) { ls[tid] = 0.0f; lq[tid] = 0.0f; }
#if ASYNC_LDS
    __builtin_amdgcn_s_wait_asynccnt(0);
#endif
    __syncthreads();

    const int wave = tid >> 5;
    const int lane = tid & 31;
    const int r16  = lane & 15;       // row (A) / col (B,C) within 16
    const int half = lane >> 4;       // K-half selector (contiguous k = 16*half + s used here)
    const int rt   = wave >> 2;       // row-tile 0/1
    const int n0   = coff + (wave & 3) * 64;

    const __bf16* arow = (const __bf16*)atile[rt * 16 + r16];
    const __bf16* wcol = (const __bf16*)Wb + (size_t)(n0 + r16) * DD + half * 16;

    auto load_a = [&](int k) -> v16bf {
        // A fragment 16x32: lane holds row r16, K chunk
        // [half*16, half*16+16).
        v16bf r;
        const v8bf c0 = *(const v8bf*)(arow + k + half * 16);
        const v8bf c1 = *(const v8bf*)(arow + k + half * 16 + 8);
#pragma unroll
        for (int i = 0; i < 8; ++i) { r[i] = c0[i]; r[i + 8] = c1[i]; }
        return r;
    };

    v8f acc[4] = {};
    v16bf acur = load_a(0);
    v16bf bcur[4], bnxt[4];
#pragma unroll
    for (int f = 0; f < 4; ++f)
        bcur[f] = *(const v16bf*)(wcol + (size_t)f * 16 * DD);

#pragma unroll
    for (int k = 0; k < DD; k += 32) {
        const int kn = k + 32;
        v16bf anxt;
        if (kn < DD) {
            anxt = load_a(kn);
#pragma unroll
            for (int f = 0; f < 4; ++f)
                bnxt[f] = *(const v16bf*)(wcol + kn + (size_t)f * 16 * DD);
        }
#pragma unroll
        for (int f = 0; f < 4; ++f)
            acc[f] = __builtin_amdgcn_wmma_f32_16x16x32_bf16(
                false, acur, false, bcur[f], (short)0, acc[f], false, false);
        if (kn < DD) {
            acur = anxt;
#pragma unroll
            for (int f = 0; f < 4; ++f) bcur[f] = bnxt[f];
        }
    }

    // C/D layout: VGPR r, lane L -> M = (L>>4)*8 + r, Ncol = L&15.
    float srow[8];
#pragma unroll
    for (int r = 0; r < 8; ++r) srow[r] = s[m0 + rt * 16 + half * 8 + r];

#pragma unroll
    for (int f = 0; f < 4; ++f) {
        const int col = n0 + f * 16 + r16;
        const float bc = bias[col];
        float sum = 0.0f, sq = 0.0f;
#pragma unroll
        for (int r = 0; r < 8; ++r) {
            const long row = m0 + rt * 16 + half * 8 + r;
            float v = acc[f][r] * srow[r] + bc;
            Y[row * DD + col] = v;
            sum += v;
            sq  += v * v;
        }
        if (do_stats) {
            atomicAdd(&ls[col - coff], sum);
            atomicAdd(&lq[col - coff], sq);
        }
    }

    if (do_stats) {
        __syncthreads();
        atomicAdd(&colsum[coff + tid], ls[tid]);
        atomicAdd(&colsq[coff + tid],  lq[tid]);
    }
}

// ---------------- BatchNorm finalize / apply ----------------

__global__ void zero_stats(float* __restrict__ colsum, float* __restrict__ colsq) {
    int c = threadIdx.x;            // launched with 512 threads
    colsum[c] = 0.0f;
    colsq[c]  = 0.0f;
}

__global__ void stats_final(const float* __restrict__ colsum, const float* __restrict__ colsq,
                            float* __restrict__ nscale, float* __restrict__ nshift,
                            float invN) {
    int c = threadIdx.x;            // launched with 512 threads
    float mu  = colsum[c] * invN;
    float var = colsq[c] * invN - mu * mu;
    float a   = rsqrtf(var + BN_EPS);
    nscale[c] = a;
    nshift[c] = -mu * a;
}

__global__ void norm_relu_cvt(const float* __restrict__ Y,
                              const float* __restrict__ nscale,
                              const float* __restrict__ nshift,
                              unsigned short* __restrict__ Xb, size_t total) {
    size_t i = (size_t)blockIdx.x * blockDim.x + threadIdx.x;
    if (i >= total) return;
    int c = (int)(i & (DD - 1));
    float v = fmaf(Y[i], nscale[c], nshift[c]);
    v = fmaxf(v, 0.0f);
    Xb[i] = f2bf(v);
}

// ---------------- host launcher ----------------

extern "C" void kernel_launch(void* const* d_in, const int* in_sizes, int n_in,
                              void* d_out, int out_size, void* d_ws, size_t ws_size,
                              hipStream_t stream) {
    const float* X  = (const float*)d_in[0];
    const int*   EI = (const int*)d_in[1];
    const float* W0 = (const float*)d_in[2];
    const float* b0 = (const float*)d_in[3];
    const float* W1 = (const float*)d_in[4];
    const float* b1 = (const float*)d_in[5];
    const float* W2 = (const float*)d_in[6];
    const float* b2 = (const float*)d_in[7];
    (void)n_in; (void)out_size; (void)ws_size;

    const int n = in_sizes[0] / DD;    // 100000 (multiple of 32)
    const int e = in_sizes[1] / 2;     // 3200000

    char* ws = (char*)d_ws;
    size_t off = 0;
    auto take = [&](size_t bytes) -> char* {
        char* p = ws + off;
        off = (off + bytes + 255) & ~(size_t)255;
        return p;
    };
    float*          deg    = (float*)take((size_t)n * 4);
    float*          s      = (float*)take((size_t)n * 4);
    float*          colsum = (float*)take(DD * 4);
    float*          colsq  = (float*)take(DD * 4);
    float*          nsc    = (float*)take(DD * 4);
    float*          nsh    = (float*)take(DD * 4);
    unsigned short* W0b    = (unsigned short*)take((size_t)DD * DD * 2);
    unsigned short* W1b    = (unsigned short*)take((size_t)DD * DD * 2);
    unsigned short* W2b    = (unsigned short*)take((size_t)DD * DD * 2);
    unsigned short* Xb     = (unsigned short*)take((size_t)n * DD * 2);
    float*          Y      = (float*)take((size_t)n * DD * 4);

    const int T = 256;
    const size_t nx = (size_t)n * DD;
    const int nxBlocks = (int)((nx + T - 1) / T);
    const int wBlocks  = (DD * DD + T - 1) / T;
    const dim3 gGrid(n / 32, 2);
    const float invN   = 1.0f / (float)n;

    // graph scale s[i]
    init_deg_s<<<(n + T - 1) / T, T, 0, stream>>>(deg, s, n);
    deg_kernel<<<(e + T - 1) / T, T, 0, stream>>>(EI, e, deg);
    wsum_kernel<<<(e + T - 1) / T, T, 0, stream>>>(EI, e, deg, s);
    s_final<<<(n + T - 1) / T, T, 0, stream>>>(s, deg, n);

    // bf16 conversions
    cvt_bf16<<<nxBlocks, T, 0, stream>>>(X, Xb, nx);
    cvt_bf16<<<wBlocks, T, 0, stream>>>(W0, W0b, (size_t)DD * DD);
    cvt_bf16<<<wBlocks, T, 0, stream>>>(W1, W1b, (size_t)DD * DD);
    cvt_bf16<<<wBlocks, T, 0, stream>>>(W2, W2b, (size_t)DD * DD);

    // layer 0: gemm*s+b -> BN -> relu -> bf16
    zero_stats<<<1, DD, 0, stream>>>(colsum, colsq);
    gcn_gemm_wmma<<<gGrid, 256, 0, stream>>>(Xb, W0b, b0, s, Y, colsum, colsq, 1);
    stats_final<<<1, DD, 0, stream>>>(colsum, colsq, nsc, nsh, invN);
    norm_relu_cvt<<<nxBlocks, T, 0, stream>>>(Y, nsc, nsh, Xb, nx);

    // layer 1
    zero_stats<<<1, DD, 0, stream>>>(colsum, colsq);
    gcn_gemm_wmma<<<gGrid, 256, 0, stream>>>(Xb, W1b, b1, s, Y, colsum, colsq, 1);
    stats_final<<<1, DD, 0, stream>>>(colsum, colsq, nsc, nsh, invN);
    norm_relu_cvt<<<nxBlocks, T, 0, stream>>>(Y, nsc, nsh, Xb, nx);

    // layer 2: gemm*s+b straight to output (f32), no BN
    gcn_gemm_wmma<<<gGrid, 256, 0, stream>>>(Xb, W2b, b2, s, (float*)d_out,
                                             colsum, colsq, 0);
}
